// CANOS_32006096290122
// MI455X (gfx1250) — compile-verified, matrix operand order checked by reading, not run on hardware
//
#include <hip/hip_runtime.h>
#include <hip/hip_bf16.h>

// ---------------------------------------------------------------------------
// CANOS power-grid GNN forward for MI455X (gfx1250).
// All per-step MLPs run as bf16 WMMA GEMMs (v_wmma_f32_16x16x32_bf16) with
// fused bias/ReLU/LayerNorm between the two matmuls. Weights are packed once
// per launch into *fragment-major* bf16 layout so every WMMA B fragment is
// two 128-bit fully-coalesced loads; A fragments are two 128-bit loads from
// row-major bf16 activations. Segment sums use f32 global atomics.
// Workspace (~111 MB) is bump-allocated from d_ws.
// ---------------------------------------------------------------------------

#define Hh      128
#define T_STEPS 48
#define NBUS    10000
#define NGEN    2000
#define NLOAD   6000
#define NSHUNT  1000
#define NLINE_  16000
#define NTRAFO_ 4000

typedef __attribute__((ext_vector_type(16))) __bf16 v16bf;
typedef __attribute__((ext_vector_type(8)))  float  v8f;

__device__ __forceinline__ unsigned short f2bf(float f) {
  unsigned int u = __float_as_uint(f);
  u += 0x7FFFu + ((u >> 16) & 1u);
  return (unsigned short)(u >> 16);
}

// --------------------------- weight packing --------------------------------
// W [steps][K][128] f32 row-major -> fragment-major packed bf16 pairs:
// out dword index = ((s*(K/32) + kbblk)*8 + t)*256 + lane*8 + v
// holding {bf16(W[k0+1][n])<<16 | bf16(W[k0][n])} with
//   n  = t*16 + (lane&15)
//   k0 = kbblk*32 + (v>=4?16:0) + (lane>>4)*8 + (v&3)*2
// so each lane's 8 fragment dwords are 32 contiguous bytes (2x b128 loads).
__global__ void pack_weights_kernel(const float* __restrict__ W,
                                    unsigned int* __restrict__ out,
                                    int K, int steps) {
  int tid = blockIdx.x * blockDim.x + threadIdx.x;
  int perStep = (K >> 5) * 2048;   // == K*64
  int total = steps * perStep;
  if (tid >= total) return;
  int s = tid / perStep;
  int rem = tid - s * perStep;
  int kbblk = rem >> 11;
  int r2 = rem & 2047;
  int t = r2 >> 8;
  int lane = (r2 >> 3) & 31;
  int v = r2 & 7;
  int n = t * 16 + (lane & 15);
  int k0 = kbblk * 32 + ((v >= 4) ? 16 : 0) + (lane >> 4) * 8 + (v & 3) * 2;
  const float* Ws = W + (size_t)s * K * Hh;
  unsigned int lo = f2bf(Ws[(size_t)k0 * Hh + n]);
  unsigned int hi = f2bf(Ws[(size_t)(k0 + 1) * Hh + n]);
  out[tid] = lo | (hi << 16);
}

__global__ void zero_f32_kernel(float* __restrict__ p, int n) {
  int i = blockIdx.x * blockDim.x + threadIdx.x;
  if (i < n) p[i] = 0.0f;
}

// ------------------------------ encoders -----------------------------------
__global__ void encoder_kernel(const float* __restrict__ x,
                               const float* __restrict__ W,
                               const float* __restrict__ b,
                               float* __restrict__ out, int rows, int din) {
  int tid = blockIdx.x * blockDim.x + threadIdx.x;
  if (tid >= rows * Hh) return;
  int row = tid >> 7;
  int j = tid & 127;
  float a = b[j];
  for (int i = 0; i < din; ++i) a += x[(size_t)row * din + i] * W[i * Hh + j];
  out[tid] = a;
}

// --------------------------- gather / concat -------------------------------
__global__ void gather_cat_kernel(const float* __restrict__ s0, const int* __restrict__ i0,
                                  const float* __restrict__ s1, const int* __restrict__ i1,
                                  const float* __restrict__ s2, const int* __restrict__ i2,
                                  int K1, int rows, int rowsPad,
                                  unsigned short* __restrict__ out) {
  int tid = blockIdx.x * blockDim.x + threadIdx.x;
  int c8 = K1 >> 3;
  int total = rowsPad * c8;
  if (tid >= total) return;
  int row = tid / c8;
  int col = (tid - row * c8) * 8;
  union { unsigned short v[8]; uint4 q; } u;
  if (row < rows) {
    int seg = col >> 7;
    int c = col & 127;
    const float* s = (seg == 0) ? s0 : ((seg == 1) ? s1 : s2);
    const int* ix  = (seg == 0) ? i0 : ((seg == 1) ? i1 : i2);
    int r = ix ? ix[row] : row;
    const float* p = s + (size_t)r * Hh + c;
#pragma unroll
    for (int j = 0; j < 8; ++j) u.v[j] = f2bf(p[j]);
  } else {
#pragma unroll
    for (int j = 0; j < 8; ++j) u.v[j] = 0;
  }
  *(uint4*)(out + (size_t)row * K1 + col) = u.q;
}

// --------------------------- fused MLP (WMMA) ------------------------------
// out = LN(relu(X@W1+b1))*g+beta @ W2 + b2 ; epilogue: write / residual /
// atomic scatter-add. Block = 256 threads (8 waves); tile 128 rows x 128 cols.
#define MODE_WRITE    0
#define MODE_RESIDUAL 1
#define MODE_SCATTER  2

__global__ __launch_bounds__(256)
void mlp_fused_kernel(const unsigned short* __restrict__ Xcat,
                      const unsigned int* __restrict__ W1p,
                      const unsigned int* __restrict__ W2p,
                      const float* __restrict__ b1,
                      const float* __restrict__ gamma,
                      const float* __restrict__ beta,
                      const float* __restrict__ b2,
                      float* __restrict__ dest,
                      const int* __restrict__ idx,
                      int rows, int K1, int mode) {
  __shared__ unsigned short ldsH[128 * 128];  // LN'd hidden, bf16, row-major

  const int lane = threadIdx.x & 31;
  const int wave = threadIdx.x >> 5;
  const int l15  = lane & 15;
  const int hif  = lane >> 4;
  const int m0   = blockIdx.x * 128 + wave * 16;

  const v8f zf = {0.f, 0.f, 0.f, 0.f, 0.f, 0.f, 0.f, 0.f};
  v8f acc[8];
#pragma unroll
  for (int t = 0; t < 8; ++t) acc[t] = zf;

  const unsigned short* xrow = Xcat + (size_t)(m0 + l15) * K1;

  // ---- GEMM1: [16 x K1] x [K1 x 128] ----
  for (int kb = 0; kb < K1; kb += 32) {
    union { v16bf v; uint4 q[2]; } a;
    const unsigned short* ap = xrow + kb + hif * 8;
    a.q[0] = *(const uint4*)(ap);
    a.q[1] = *(const uint4*)(ap + 16);
    const unsigned int* wb = W1p + (size_t)(kb >> 5) * 2048 + lane * 8;
#pragma unroll
    for (int t = 0; t < 8; ++t) {
      union { v16bf v; uint4 q[2]; } b;
      const uint4* bp = (const uint4*)(wb + t * 256);
      b.q[0] = bp[0];
      b.q[1] = bp[1];
      acc[t] = __builtin_amdgcn_wmma_f32_16x16x32_bf16(
          false, a.v, false, b.v, (short)0, acc[t], false, false);
    }
  }

  // ---- bias + relu + LayerNorm (per row of 128, all in-wave) ----
  float rmean[8], rrstd[8];
  {
    float rsum[8], rsq[8];
#pragma unroll
    for (int e = 0; e < 8; ++e) { rsum[e] = 0.f; rsq[e] = 0.f; }
#pragma unroll
    for (int t = 0; t < 8; ++t) {
      int n = t * 16 + l15;
      float bb = b1[n];
#pragma unroll
      for (int e = 0; e < 8; ++e) {
        float hv = acc[t][e] + bb;
        hv = fmaxf(hv, 0.f);
        acc[t][e] = hv;
        rsum[e] += hv;
        rsq[e]  += hv * hv;
      }
    }
#pragma unroll
    for (int e = 0; e < 8; ++e) {
      float s = rsum[e], q = rsq[e];
      for (int msk = 1; msk < 16; msk <<= 1) {
        s += __shfl_xor(s, msk, 32);
        q += __shfl_xor(q, msk, 32);
      }
      float mean = s * (1.f / 128.f);
      float var  = q * (1.f / 128.f) - mean * mean;
      rmean[e] = mean;
      rrstd[e] = rsqrtf(var + 1e-5f);
    }
  }
#pragma unroll
  for (int t = 0; t < 8; ++t) {
    int n = t * 16 + l15;
    float gg = gamma[n], be = beta[n];
#pragma unroll
    for (int e = 0; e < 8; ++e) {
      float hn = (acc[t][e] - rmean[e]) * rrstd[e] * gg + be;
      int m = wave * 16 + e + hif * 8;
      ldsH[m * 128 + n] = f2bf(hn);
    }
  }
  __syncthreads();

  // ---- GEMM2: [16 x 128] x [128 x 128] ----
  v8f acc2[8];
#pragma unroll
  for (int t = 0; t < 8; ++t) acc2[t] = zf;
  const unsigned short* hrow = &ldsH[(wave * 16 + l15) * 128];
  for (int kb = 0; kb < 128; kb += 32) {
    union { v16bf v; uint4 q[2]; } a;
    const unsigned short* ap = hrow + kb + hif * 8;
    a.q[0] = *(const uint4*)(ap);           // ds_load_b128
    a.q[1] = *(const uint4*)(ap + 16);
    const unsigned int* wb = W2p + (size_t)(kb >> 5) * 2048 + lane * 8;
#pragma unroll
    for (int t = 0; t < 8; ++t) {
      union { v16bf v; uint4 q[2]; } b;
      const uint4* bp = (const uint4*)(wb + t * 256);
      b.q[0] = bp[0];
      b.q[1] = bp[1];
      acc2[t] = __builtin_amdgcn_wmma_f32_16x16x32_bf16(
          false, a.v, false, b.v, (short)0, acc2[t], false, false);
    }
  }

  // ---- epilogue ----
#pragma unroll
  for (int t = 0; t < 8; ++t) {
    int n = t * 16 + l15;
    float bb = b2[n];
#pragma unroll
    for (int e = 0; e < 8; ++e) {
      int grow = m0 + e + hif * 8;
      if (grow < rows) {
        float o = acc2[t][e] + bb;
        if (mode == MODE_SCATTER) {
          int d = idx[grow];
          atomicAdd(dest + (size_t)d * Hh + n, o);
        } else if (mode == MODE_RESIDUAL) {
          dest[(size_t)grow * Hh + n] += o;
        } else {
          dest[(size_t)grow * Hh + n] = o;
        }
      }
    }
  }
}

// ------------------------------- decoder -----------------------------------
__global__ __launch_bounds__(256)
void decoder_kernel(const float* __restrict__ h,
                    const float* __restrict__ W1, const float* __restrict__ b1,
                    const float* __restrict__ gamma, const float* __restrict__ beta,
                    const float* __restrict__ W2, const float* __restrict__ b2,
                    const float* __restrict__ W3, const float* __restrict__ b3,
                    const float* __restrict__ lo0, const float* __restrict__ hi0, int sig0,
                    const float* __restrict__ lo1, const float* __restrict__ hi1,
                    float* __restrict__ out0, float* __restrict__ out1, int rows) {
  __shared__ float xs[128];
  __shared__ float hs[256];
  __shared__ float red[256];
  int row = blockIdx.x;
  if (row >= rows) return;
  int j = threadIdx.x;
  if (j < 128) xs[j] = h[(size_t)row * Hh + j];
  __syncthreads();
  float a = b1[j];
  for (int i = 0; i < 128; ++i) a += xs[i] * W1[i * 256 + j];
  a = fmaxf(a, 0.f);
  red[j] = a; __syncthreads();
  for (int s = 128; s > 0; s >>= 1) { if (j < s) red[j] += red[j + s]; __syncthreads(); }
  float mean = red[0] * (1.f / 256.f); __syncthreads();
  float d = a - mean;
  red[j] = d * d; __syncthreads();
  for (int s = 128; s > 0; s >>= 1) { if (j < s) red[j] += red[j + s]; __syncthreads(); }
  float var = red[0] * (1.f / 256.f); __syncthreads();
  float hn = d * rsqrtf(var + 1e-5f) * gamma[j] + beta[j];
  hs[j] = hn; __syncthreads();
  float o = b2[j];
  for (int i = 0; i < 256; ++i) o += hs[i] * W2[i * 256 + j];
  red[j] = o * W3[j * 2 + 0]; __syncthreads();
  for (int s = 128; s > 0; s >>= 1) { if (j < s) red[j] += red[j + s]; __syncthreads(); }
  float y0 = red[0] + b3[0]; __syncthreads();
  red[j] = o * W3[j * 2 + 1]; __syncthreads();
  for (int s = 128; s > 0; s >>= 1) { if (j < s) red[j] += red[j + s]; __syncthreads(); }
  float y1 = red[0] + b3[1];
  if (j == 0) {
    float r0;
    if (sig0) { float l = lo0[row], hh = hi0[row]; r0 = l + (hh - l) / (1.f + expf(-y0)); }
    else r0 = y0;
    float l1 = lo1[row], h1 = hi1[row];
    out0[row] = r0;
    out1[row] = l1 + (h1 - l1) / (1.f + expf(-y1));
  }
}

// ------------------------- AC power-flow tails -----------------------------
__global__ void pf_line_kernel(const float* __restrict__ va, const float* __restrict__ vm,
                               const float* __restrict__ lx,
                               const int* __restrict__ lf, const int* __restrict__ lt,
                               float* __restrict__ sfre, float* __restrict__ sfim,
                               float* __restrict__ stre, float* __restrict__ stim, int n) {
  int i = blockIdx.x * blockDim.x + threadIdx.x;
  if (i >= n) return;
  float bcf = lx[i * 6 + 2], bct = lx[i * 6 + 3], r = lx[i * 6 + 4], x = lx[i * 6 + 5];
  float den = r * r + x * x;
  float gy = r / den, by = -x / den;       // y_l = gy + j*by
  int f = lf[i], t = lt[i];
  float vmf = vm[f], vaf = va[f], vmt = vm[t], vat = va[t];
  float af2 = vmf * vmf, at2 = vmt * vmt, mm = vmf * vmt;
  float dd = vaf - vat;
  float cd = cosf(dd), sd = sinf(dd);
  float pr = mm * cd, pi = mm * sd;
  sfre[i] = gy * af2 - (gy * pr + by * pi);
  sfim[i] = -(by + bcf) * af2 - (gy * pi - by * pr);
  float qr = mm * cd, qi = -mm * sd;
  stre[i] = gy * at2 - (gy * qr + by * qi);
  stim[i] = -(by + bct) * at2 - (gy * qi - by * qr);
}

__global__ void pf_trafo_kernel(const float* __restrict__ va, const float* __restrict__ vm,
                                const float* __restrict__ tx,
                                const int* __restrict__ tf, const int* __restrict__ tt,
                                float* __restrict__ sfre, float* __restrict__ sfim,
                                float* __restrict__ stre, float* __restrict__ stim, int n) {
  int i = blockIdx.x * blockDim.x + threadIdx.x;
  if (i >= n) return;
  float bcf = tx[i * 11 + 2], bct = tx[i * 11 + 3];
  float r = tx[i * 11 + 4], x = tx[i * 11 + 5];
  float tap = fmaxf(tx[i * 11 + 9], 1e-4f);
  float shift = tx[i * 11 + 10];
  float den = r * r + x * x;
  float gy = r / den, by = -x / den;
  int f = tf[i], t = tt[i];
  float vmf = vm[f], vaf = va[f], vmt = vm[t], vat = va[t];
  float af2 = vmf * vmf, at2 = vmt * vmt, mm = vmf * vmt;
  float inv_tap = 1.f / tap, inv_tap2 = inv_tap * inv_tap;
  {
    float ang = vaf - vat - shift;
    float pr = mm * cosf(ang) * inv_tap, pi = mm * sinf(ang) * inv_tap;
    sfre[i] = gy * af2 * inv_tap2 - (gy * pr + by * pi);
    sfim[i] = -(by + bcf) * af2 * inv_tap2 - (gy * pi - by * pr);
  }
  {
    float ang = vat - vaf + shift;
    float qr = mm * cosf(ang) * inv_tap, qi = mm * sinf(ang) * inv_tap;
    stre[i] = gy * at2 - (gy * qr + by * qi);
    stim[i] = -(by + bct) * at2 - (gy * qi - by * qr);
  }
}

// ---------------------------------------------------------------------------
extern "C" void kernel_launch(void* const* d_in, const int* in_sizes, int n_in,
                              void* d_out, int out_size, void* d_ws, size_t ws_size,
                              hipStream_t stream) {
  (void)in_sizes; (void)n_in; (void)out_size; (void)ws_size;

  const float* bus_x   = (const float*)d_in[0];
  const float* gen_x   = (const float*)d_in[1];
  const float* load_x  = (const float*)d_in[2];
  const float* shunt_x = (const float*)d_in[3];
  const float* line_x  = (const float*)d_in[4];
  const float* trafo_x = (const float*)d_in[5];
  const int* line_from  = (const int*)d_in[6];
  const int* line_to    = (const int*)d_in[7];
  const int* trafo_from = (const int*)d_in[8];
  const int* trafo_to   = (const int*)d_in[9];
  const int* gen_bus    = (const int*)d_in[10];
  const int* load_bus   = (const int*)d_in[11];
  const int* shunt_bus  = (const int*)d_in[12];
  const float* bus_vm_min = (const float*)d_in[13];
  const float* bus_vm_max = (const float*)d_in[14];
  const float* gen_pg_min = (const float*)d_in[15];
  const float* gen_pg_max = (const float*)d_in[16];
  const float* gen_qg_min = (const float*)d_in[17];
  const float* gen_qg_max = (const float*)d_in[18];
  const float* prm        = (const float*)d_in[19];
  float* out = (float*)d_out;

  // ---- parameter blob offsets (jax pytree: dicts flattened sorted by key) --
  size_t off = 0;
  auto take = [&](size_t nel) { size_t o = off; off += nel; return o; };
  struct MlpOff { size_t W1, W2, b1, b2, beta, g; int din; };
  struct DecOff { size_t W1, W2, W3, b1, b2, b3, beta, g; };
  struct LinOff { size_t W, b; int din; };
  auto mlp_off = [&](int din) {
    MlpOff m; m.din = din;
    m.W1 = take((size_t)T_STEPS * din * Hh);
    m.W2 = take((size_t)T_STEPS * Hh * Hh);
    m.b1 = take((size_t)T_STEPS * Hh);
    m.b2 = take((size_t)T_STEPS * Hh);
    m.beta = take((size_t)T_STEPS * Hh);
    m.g = take((size_t)T_STEPS * Hh);
    return m;
  };
  auto dec_off = [&]() {
    DecOff d;
    d.W1 = take(128 * 256); d.W2 = take(256 * 256); d.W3 = take(256 * 2);
    d.b1 = take(256); d.b2 = take(256); d.b3 = take(2);
    d.beta = take(256); d.g = take(256);
    return d;
  };
  auto lin_off = [&](int din) { LinOff l; l.din = din; l.W = take((size_t)din * Hh); l.b = take(Hh); return l; };

  MlpOff m_bus_upd   = mlp_off(256);
  DecOff dec_bus     = dec_off();
  DecOff dec_gen     = dec_off();
  LinOff enc_bus     = lin_off(4);
  LinOff enc_gen     = lin_off(8);
  LinOff enc_line    = lin_off(6);
  LinOff enc_load    = lin_off(2);
  LinOff enc_shunt   = lin_off(2);
  LinOff enc_trafo   = lin_off(11);
  MlpOff m_gen_upd   = mlp_off(256);
  MlpOff m_line_edge = mlp_off(384);
  MlpOff m_load_upd  = mlp_off(256);
  MlpOff m_msg_bus_gen   = mlp_off(256);
  MlpOff m_msg_bus_line  = mlp_off(256);
  MlpOff m_msg_bus_load  = mlp_off(256);
  MlpOff m_msg_bus_shunt = mlp_off(256);
  MlpOff m_msg_bus_trafo = mlp_off(256);
  MlpOff m_msg_gen_bus   = mlp_off(256);
  MlpOff m_msg_load_bus  = mlp_off(256);
  MlpOff m_msg_shunt_bus = mlp_off(256);
  MlpOff m_shunt_upd = mlp_off(256);
  MlpOff m_trafo_edge = mlp_off(384);

  const MlpOff mlps[14] = {
    m_bus_upd, m_gen_upd, m_line_edge, m_load_upd, m_msg_bus_gen, m_msg_bus_line,
    m_msg_bus_load, m_msg_bus_shunt, m_msg_bus_trafo, m_msg_gen_bus,
    m_msg_load_bus, m_msg_shunt_bus, m_shunt_upd, m_trafo_edge };
  enum { M_BUS_UPD = 0, M_GEN_UPD, M_LINE_EDGE, M_LOAD_UPD, M_MSG_BUS_GEN,
         M_MSG_BUS_LINE, M_MSG_BUS_LOAD, M_MSG_BUS_SHUNT, M_MSG_BUS_TRAFO,
         M_MSG_GEN_BUS, M_MSG_LOAD_BUS, M_MSG_SHUNT_BUS, M_SHUNT_UPD, M_TRAFO_EDGE };

  // ---- workspace bump allocator ----
  char* wsp = (char*)d_ws;
  size_t woff = 0;
  auto walloc = [&](size_t bytes) {
    size_t o = (woff + 255) & ~(size_t)255;
    woff = o + bytes;
    return (void*)(wsp + o);
  };

  float* hb    = (float*)walloc((size_t)NBUS * Hh * 4);
  float* hg    = (float*)walloc((size_t)NGEN * Hh * 4);
  float* hl    = (float*)walloc((size_t)NLOAD * Hh * 4);
  float* hsv   = (float*)walloc((size_t)NSHUNT * Hh * 4);
  float* hline = (float*)walloc((size_t)NLINE_ * Hh * 4);
  float* htr   = (float*)walloc((size_t)NTRAFO_ * Hh * 4);
  float* bus_agg   = (float*)walloc((size_t)NBUS * Hh * 4);
  float* gen_agg   = (float*)walloc((size_t)NGEN * Hh * 4);
  float* load_agg  = (float*)walloc((size_t)NLOAD * Hh * 4);
  float* shunt_agg = (float*)walloc((size_t)NSHUNT * Hh * 4);
  unsigned short* xcat = (unsigned short*)walloc((size_t)NLINE_ * 384 * 2);

  unsigned int* W1pk[14];
  unsigned int* W2pk[14];
  for (int i = 0; i < 14; ++i) {
    int din = mlps[i].din;
    W1pk[i] = (unsigned int*)walloc((size_t)T_STEPS * (din / 2) * 128 * 4);
    W2pk[i] = (unsigned int*)walloc((size_t)T_STEPS * 64 * 128 * 4);
  }

  // ---- pack all step-MLP weights to fragment-major bf16 layout ----
  for (int i = 0; i < 14; ++i) {
    int din = mlps[i].din;
    int tot1 = T_STEPS * (din / 2) * 128;
    pack_weights_kernel<<<(tot1 + 255) / 256, 256, 0, stream>>>(
        prm + mlps[i].W1, W1pk[i], din, T_STEPS);
    int tot2 = T_STEPS * 64 * 128;
    pack_weights_kernel<<<(tot2 + 255) / 256, 256, 0, stream>>>(
        prm + mlps[i].W2, W2pk[i], 128, T_STEPS);
  }

  // ---- encoders ----
  auto enc = [&](const float* x, const LinOff& L, float* dst, int rows) {
    int tot = rows * Hh;
    encoder_kernel<<<(tot + 255) / 256, 256, 0, stream>>>(
        x, prm + L.W, prm + L.b, dst, rows, L.din);
  };
  enc(bus_x, enc_bus, hb, NBUS);
  enc(gen_x, enc_gen, hg, NGEN);
  enc(load_x, enc_load, hl, NLOAD);
  enc(shunt_x, enc_shunt, hsv, NSHUNT);
  enc(line_x, enc_line, hline, NLINE_);
  enc(trafo_x, enc_trafo, htr, NTRAFO_);

  auto pad128 = [](int r) { return (r + 127) & ~127; };
  auto gather = [&](const float* s0, const int* i0, const float* s1, const int* i1,
                    const float* s2, const int* i2, int K1, int rows) {
    int rowsPad = pad128(rows);
    int tot = rowsPad * (K1 >> 3);
    gather_cat_kernel<<<(tot + 255) / 256, 256, 0, stream>>>(
        s0, i0, s1, i1, s2, i2, K1, rows, rowsPad, xcat);
  };
  auto run_mlp = [&](int id, int t, float* dest, const int* idx, int rows, int mode) {
    const MlpOff& M = mlps[id];
    int K1 = M.din;
    const unsigned int* w1 = W1pk[id] + (size_t)t * (K1 / 2) * 128;
    const unsigned int* w2 = W2pk[id] + (size_t)t * 64 * 128;
    int blocks = pad128(rows) / 128;
    mlp_fused_kernel<<<blocks, 256, 0, stream>>>(
        xcat, w1, w2,
        prm + M.b1 + (size_t)t * Hh, prm + M.g + (size_t)t * Hh,
        prm + M.beta + (size_t)t * Hh, prm + M.b2 + (size_t)t * Hh,
        dest, idx, rows, K1, mode);
  };

  // ---- 48 message-passing steps ----
  for (int t = 0; t < T_STEPS; ++t) {
    gather(hb, line_from, hb, line_to, hline, nullptr, 384, NLINE_);
    run_mlp(M_LINE_EDGE, t, hline, nullptr, NLINE_, MODE_RESIDUAL);
    gather(hb, trafo_from, hb, trafo_to, htr, nullptr, 384, NTRAFO_);
    run_mlp(M_TRAFO_EDGE, t, htr, nullptr, NTRAFO_, MODE_RESIDUAL);

    zero_f32_kernel<<<(NBUS * Hh + 255) / 256, 256, 0, stream>>>(bus_agg, NBUS * Hh);

    gather(hb, line_from, hline, nullptr, nullptr, nullptr, 256, NLINE_);
    run_mlp(M_MSG_BUS_LINE, t, bus_agg, line_to, NLINE_, MODE_SCATTER);
    gather(hb, line_to, hline, nullptr, nullptr, nullptr, 256, NLINE_);
    run_mlp(M_MSG_BUS_LINE, t, bus_agg, line_from, NLINE_, MODE_SCATTER);
    gather(hb, trafo_from, htr, nullptr, nullptr, nullptr, 256, NTRAFO_);
    run_mlp(M_MSG_BUS_TRAFO, t, bus_agg, trafo_to, NTRAFO_, MODE_SCATTER);
    gather(hb, trafo_to, htr, nullptr, nullptr, nullptr, 256, NTRAFO_);
    run_mlp(M_MSG_BUS_TRAFO, t, bus_agg, trafo_from, NTRAFO_, MODE_SCATTER);
    gather(hg, nullptr, hb, gen_bus, nullptr, nullptr, 256, NGEN);
    run_mlp(M_MSG_BUS_GEN, t, bus_agg, gen_bus, NGEN, MODE_SCATTER);
    gather(hl, nullptr, hb, load_bus, nullptr, nullptr, 256, NLOAD);
    run_mlp(M_MSG_BUS_LOAD, t, bus_agg, load_bus, NLOAD, MODE_SCATTER);
    gather(hsv, nullptr, hb, shunt_bus, nullptr, nullptr, 256, NSHUNT);
    run_mlp(M_MSG_BUS_SHUNT, t, bus_agg, shunt_bus, NSHUNT, MODE_SCATTER);
    gather(hb, gen_bus, hg, nullptr, nullptr, nullptr, 256, NGEN);
    run_mlp(M_MSG_GEN_BUS, t, gen_agg, nullptr, NGEN, MODE_WRITE);
    gather(hb, load_bus, hl, nullptr, nullptr, nullptr, 256, NLOAD);
    run_mlp(M_MSG_LOAD_BUS, t, load_agg, nullptr, NLOAD, MODE_WRITE);
    gather(hb, shunt_bus, hsv, nullptr, nullptr, nullptr, 256, NSHUNT);
    run_mlp(M_MSG_SHUNT_BUS, t, shunt_agg, nullptr, NSHUNT, MODE_WRITE);

    gather(hb, nullptr, bus_agg, nullptr, nullptr, nullptr, 256, NBUS);
    run_mlp(M_BUS_UPD, t, hb, nullptr, NBUS, MODE_RESIDUAL);
    gather(hg, nullptr, gen_agg, nullptr, nullptr, nullptr, 256, NGEN);
    run_mlp(M_GEN_UPD, t, hg, nullptr, NGEN, MODE_RESIDUAL);
    gather(hl, nullptr, load_agg, nullptr, nullptr, nullptr, 256, NLOAD);
    run_mlp(M_LOAD_UPD, t, hl, nullptr, NLOAD, MODE_RESIDUAL);
    gather(hsv, nullptr, shunt_agg, nullptr, nullptr, nullptr, 256, NSHUNT);
    run_mlp(M_SHUNT_UPD, t, hsv, nullptr, NSHUNT, MODE_RESIDUAL);
  }

  // ---- decoders ----
  const size_t o_va = 0, o_vm = NBUS, o_pg = 2 * (size_t)NBUS,
               o_qg = 2 * (size_t)NBUS + NGEN;
  const size_t o_sf = 2 * (size_t)NBUS + 2 * NGEN;
  const size_t o_tf = o_sf + 4 * (size_t)NLINE_;
  decoder_kernel<<<NBUS, 256, 0, stream>>>(
      hb, prm + dec_bus.W1, prm + dec_bus.b1, prm + dec_bus.g, prm + dec_bus.beta,
      prm + dec_bus.W2, prm + dec_bus.b2, prm + dec_bus.W3, prm + dec_bus.b3,
      bus_vm_min, bus_vm_max, 0, bus_vm_min, bus_vm_max,
      out + o_va, out + o_vm, NBUS);
  decoder_kernel<<<NGEN, 256, 0, stream>>>(
      hg, prm + dec_gen.W1, prm + dec_gen.b1, prm + dec_gen.g, prm + dec_gen.beta,
      prm + dec_gen.W2, prm + dec_gen.b2, prm + dec_gen.W3, prm + dec_gen.b3,
      gen_pg_min, gen_pg_max, 1, gen_qg_min, gen_qg_max,
      out + o_pg, out + o_qg, NGEN);

  // ---- AC power flow ----
  pf_line_kernel<<<(NLINE_ + 255) / 256, 256, 0, stream>>>(
      out + o_va, out + o_vm, line_x, line_from, line_to,
      out + o_sf, out + o_sf + NLINE_, out + o_sf + 2 * NLINE_, out + o_sf + 3 * NLINE_,
      NLINE_);
  pf_trafo_kernel<<<(NTRAFO_ + 255) / 256, 256, 0, stream>>>(
      out + o_va, out + o_vm, trafo_x, trafo_from, trafo_to,
      out + o_tf, out + o_tf + NTRAFO_, out + o_tf + 2 * NTRAFO_, out + o_tf + 3 * NTRAFO_,
      NTRAFO_);
}
